// SparseAttention_54039278518802
// MI455X (gfx1250) — matrix operand order repaired and checked by reading, not compiled, not run on hardware
//
#include <hip/hip_runtime.h>

typedef __attribute__((ext_vector_type(16))) __bf16 v16bf;
typedef __attribute__((ext_vector_type(8)))  float  v8f;
typedef unsigned short __attribute__((address_space(3))) lds_us_t;

constexpr int B_ = 4;
constexpr int S_ = 2048;
constexpr int E_ = 256;
constexpr int H_ = 8;
constexpr int D_ = 32;
constexpr int KSEL = 204;          // int(S * 0.1)
constexpr int WAVES = 4;
constexpr int SEG = S_ / WAVES;    // 512 columns per wave

// ---------- helpers ----------

__device__ inline unsigned short f32_to_bf16(float f) {
  unsigned u = __float_as_uint(f);
  unsigned r = u + 0x7FFFu + ((u >> 16) & 1u);   // round-to-nearest-even
  return (unsigned short)(r >> 16);
}

__device__ inline v8f v8f_zero() {
  v8f c = {0.f, 0.f, 0.f, 0.f, 0.f, 0.f, 0.f, 0.f};
  return c;
}

// A tile 16x32 bf16: value(m,k) = src[(row0+m)*rs + k0 + k].
// Per lane (M = lane&15, half = lane>>4) the ISA striping is two contiguous
// 16B runs: elements [8*hf .. 8*hf+7] and [16+8*hf .. 23+8*hf]  -> two b128 loads.
__device__ inline v16bf load_tile_A(const unsigned short* src, int row0, int rs, int k0) {
  union { uint4 q[2]; v16bf v; } t;
  const int lane = threadIdx.x & 31;
  const int r = row0 + (lane & 15);
  const int hf = lane >> 4;
  const uint4* p = (const uint4*)(src + (size_t)r * rs + k0);
  t.q[0] = p[hf];
  t.q[1] = p[2 + hf];
  return t.v;
}

// B tile 32x16 bf16: value(k,n) = src[(col0+n)*rs + k0 + k] ("N-major" source).
// Per lane (N = lane&15) striping is elements [8*hf .. 8*hf+7] and [16+8*hf .. 23+8*hf] -> two b128 loads.
__device__ inline v16bf load_tile_B(const unsigned short* src, int col0, int rs, int k0) {
  union { uint4 q[2]; v16bf v; } t;
  const int lane = threadIdx.x & 31;
  const int n = col0 + (lane & 15);
  const int hf = lane >> 4;
  const uint4* p = (const uint4*)(src + (size_t)n * rs + k0);
  t.q[0] = p[hf];
  t.q[1] = p[2 + hf];
  return t.v;
}

__device__ inline v8f wmma_bf16(v16bf a, v16bf b, v8f c) {
  return __builtin_amdgcn_wmma_f32_16x16x32_bf16(false, a, false, b, (short)0, c, false, false);
}

// Async-copy one 16x32 bf16 K tile (1 KB) global -> LDS using the gfx1250 async path.
// 2 x global_load_async_to_lds_b128: each lane moves 16B; instruction h covers rows 8h..8h+7.
__device__ inline void async_copy_ktile(const unsigned short* gsrc_base,  // &K[row0*E + kcol0]
                                        unsigned short* lds_tile) {
  const int lane = threadIdx.x & 31;
#pragma unroll
  for (int half = 0; half < 2; ++half) {
    const int row = half * 8 + (lane >> 2);
    const int chunk = (lane & 3) * 8;                       // element offset in row (16B chunks)
    const unsigned short* g = gsrc_base + (size_t)row * E_ + chunk;
    const unsigned lds_addr = (unsigned)(size_t)(lds_us_t*)(lds_tile + row * 32 + chunk);
    const unsigned long long ga = (unsigned long long)(size_t)g;
    asm volatile("global_load_async_to_lds_b128 %0, %1, off"
                 :: "v"(lds_addr), "v"(ga)
                 : "memory");
  }
}

// ---------- kernels ----------

__global__ __launch_bounds__(256) void cvt_kernel(const float* __restrict__ src,
                                                  unsigned short* __restrict__ dst, int n) {
  int i = blockIdx.x * 256 + threadIdx.x;
  if (i < n) dst[i] = f32_to_bf16(src[i]);
}

// scores[b,s,t] = (x . x^T) / 16 ; one wave per 16x16 tile; written into d_out attn region
__global__ __launch_bounds__(32) void scores_kernel(const unsigned short* __restrict__ xb,
                                                    float* __restrict__ attn) {
  const int bI = blockIdx.z;
  const int s0 = blockIdx.x * 16;
  const int t0 = blockIdx.y * 16;
  v8f c = v8f_zero();
  for (int k0 = 0; k0 < E_; k0 += 32) {
    v16bf a  = load_tile_A(xb, bI * S_ + s0, E_, k0);
    v16bf bt = load_tile_B(xb, bI * S_ + t0, E_, k0);
    c = wmma_bf16(a, bt, c);
  }
  const int lane = threadIdx.x & 31;
  const int hf = lane >> 4, colL = lane & 15;
#pragma unroll
  for (int g = 0; g < 8; ++g)
    attn[((size_t)(bI * S_ + s0 + g + 8 * hf)) * S_ + t0 + colL] = c[g] * 0.0625f;
}

// exact k-th largest per row: in-LDS bitonic sort (descending), thresh = a[KSEL-1]
__global__ __launch_bounds__(256) void thresh_kernel(const float* __restrict__ attn,
                                                     float* __restrict__ thresh) {
  __shared__ float a[S_];
  const float* src = attn + (size_t)blockIdx.x * S_;
  for (int i = threadIdx.x; i < S_; i += 256) a[i] = src[i];
  __syncthreads();
  for (int k = 2; k <= S_; k <<= 1) {
    for (int j = k >> 1; j > 0; j >>= 1) {
      for (int i = threadIdx.x; i < S_; i += 256) {
        int ixj = i ^ j;
        if (ixj > i) {
          bool desc = ((i & k) == 0);
          float x = a[i], y = a[ixj];
          if (desc ? (x < y) : (x > y)) { a[i] = y; a[ixj] = x; }
        }
      }
      __syncthreads();
    }
  }
  if (threadIdx.x == 0) thresh[blockIdx.x] = a[KSEL - 1];
}

// pack mask bits: bit(t) = scores[row,t] >= thresh[row]
__global__ __launch_bounds__(64) void mask_kernel(const float* __restrict__ attn,
                                                  const float* __restrict__ thresh,
                                                  unsigned* __restrict__ maskw) {
  const size_t row = blockIdx.x;
  const float th = thresh[row];
  const float* src = attn + row * (size_t)S_;
  const int w = threadIdx.x;  // one 32-bit word
  unsigned bits = 0u;
#pragma unroll 4
  for (int i = 0; i < 32; ++i)
    if (src[w * 32 + i] >= th) bits |= (1u << i);
  maskw[row * (S_ / 32) + w] = bits;
}

// out(n,e) = src(n,:) . W(e,:) + bias(e).  mode 0/1: bf16 row-major (Q/K);
// mode 2: bf16 transposed [b*E + e][t] (V for PV B-operand); mode 3: f32 row-major (final out)
__global__ __launch_bounds__(32) void proj_kernel(const unsigned short* __restrict__ src,
                                                  const unsigned short* __restrict__ wb,
                                                  const float* __restrict__ bias,
                                                  void* __restrict__ dst, int mode) {
  const int n0 = blockIdx.x * 16;
  const int e0 = blockIdx.y * 16;
  v8f c = v8f_zero();
  for (int k0 = 0; k0 < E_; k0 += 32) {
    v16bf a  = load_tile_A(src, n0, E_, k0);
    v16bf bt = load_tile_B(wb, e0, E_, k0);  // value(k,n) = W[e0+n][k]
    c = wmma_bf16(a, bt, c);
  }
  const int lane = threadIdx.x & 31;
  const int hf = lane >> 4, colL = lane & 15;
  const int e = e0 + colL;
  const float bv = bias[e];
#pragma unroll
  for (int g = 0; g < 8; ++g) {
    const int row = n0 + g + 8 * hf;
    const float v = c[g] + bv;
    if (mode == 0 || mode == 1) {
      ((unsigned short*)dst)[(size_t)row * E_ + e] = f32_to_bf16(v);
    } else if (mode == 2) {
      const int b = row >> 11;           // row / S_
      const int t = row & (S_ - 1);
      ((unsigned short*)dst)[((size_t)(b * E_ + e)) * S_ + t] = f32_to_bf16(v);
    } else {
      ((float*)dst)[(size_t)row * E_ + e] = v;
    }
  }
}

// attention: 1 WG (4 waves) per (b, 16-row tile); head loop inside (race-free prob accumulation)
__global__ __launch_bounds__(128) void attn_kernel(const unsigned short* __restrict__ qb,
                                                   const unsigned short* __restrict__ kb,
                                                   const unsigned short* __restrict__ vtb,
                                                   const unsigned* __restrict__ maskw,
                                                   float* __restrict__ attn,
                                                   unsigned short* __restrict__ ctxb) {
  __shared__ unsigned short Klds[WAVES][2][16 * 32];   // async double-buffered K tiles (8 KB)
  __shared__ unsigned short Pstage[WAVES][16][32];     // bf16 prob tiles (A-operand staging)
  __shared__ float mred[WAVES][16], sred[WAVES][16];
  __shared__ float Mrow[16], Srow[16];
  __shared__ float cred[WAVES][16][32];

  const int b = blockIdx.y;
  const int s0 = blockIdx.x * 16;
  const int wave = threadIdx.x >> 5;
  const int lane = threadIdx.x & 31;
  const int hf = lane >> 4;
  const int colL = lane & 15;
  const int tbase = wave * SEG;
  const float scale = 0.17677669529663687f;            // 1/sqrt(D)
  const float NEGINF = -__builtin_inff();

  for (int h = 0; h < H_; ++h) {
    const v16bf aq = load_tile_A(qb, b * S_ + s0, E_, h * D_);

    // ---- phase 1: streaming row-max / exp-sum over this wave's 512 columns,
    //      K tiles staged through LDS with async b128 copies (double-buffered)
    float m[8], s[8];
#pragma unroll
    for (int g = 0; g < 8; ++g) { m[g] = NEGINF; s[g] = 0.f; }

    async_copy_ktile(kb + ((size_t)(b * S_ + tbase) * E_ + h * D_), &Klds[wave][0][0]);
    for (int j = 0; j < SEG / 16; ++j) {
      if (j + 1 < SEG / 16) {
        async_copy_ktile(kb + ((size_t)(b * S_ + tbase + (j + 1) * 16) * E_ + h * D_),
                         &Klds[wave][(j + 1) & 1][0]);
        asm volatile("s_wait_asynccnt 0x2" ::: "memory");  // tile j landed (in-order)
      } else {
        asm volatile("s_wait_asynccnt 0x0" ::: "memory");
      }
      const int t0 = tbase + j * 16;
      v16bf bt = load_tile_B(&Klds[wave][j & 1][0], 0, 32, 0);
      v8f c = wmma_bf16(aq, bt, v8f_zero());
#pragma unroll
      for (int g = 0; g < 8; ++g) {
        const int row = s0 + g + 8 * hf;
        const int col = t0 + colL;
        const unsigned bits = maskw[(b * S_ + row) * (S_ / 32) + (col >> 5)];
        const float l = ((bits >> (col & 31)) & 1u) ? c[g] * scale : NEGINF;
        if (l != NEGINF) {
          const float nm = fmaxf(m[g], l);
          s[g] = s[g] * __expf(m[g] - nm) + __expf(l - nm);
          m[g] = nm;
        }
      }
    }
    // reduce across the 16 lanes sharing each row (xor masks stay within the half)
#pragma unroll
    for (int off = 1; off < 16; off <<= 1) {
#pragma unroll
      for (int g = 0; g < 8; ++g) {
        const float m2 = __shfl_xor(m[g], off, 32);
        const float s2 = __shfl_xor(s[g], off, 32);
        const float nm = fmaxf(m[g], m2);
        if (nm != NEGINF) {
          s[g] = s[g] * __expf(m[g] - nm) + s2 * __expf(m2 - nm);
          m[g] = nm;
        }
      }
    }
    if (colL == 0) {
#pragma unroll
      for (int g = 0; g < 8; ++g) {
        mred[wave][g + 8 * hf] = m[g];
        sred[wave][g + 8 * hf] = s[g];
      }
    }
    __syncthreads();
    if (threadIdx.x < 16) {
      float M = NEGINF, Ss = 0.f;
#pragma unroll
      for (int w = 0; w < WAVES; ++w) {
        const float mw = mred[w][threadIdx.x];
        const float sw = sred[w][threadIdx.x];
        if (mw != NEGINF) {
          const float nm = fmaxf(M, mw);
          Ss = Ss * __expf(M - nm) + sw * __expf(mw - nm);
          M = nm;
        }
      }
      Mrow[threadIdx.x] = M;
      Srow[threadIdx.x] = (Ss > 0.f) ? Ss : 1.f;
    }
    __syncthreads();

    // ---- phase 2+3 fused: recompute logits, emit probs, immediately consume as PV A-operand
    v8f c0 = v8f_zero(), c1 = v8f_zero();
    for (int ch = 0; ch < SEG / 32; ++ch) {
#pragma unroll
      for (int sub = 0; sub < 2; ++sub) {
        const int t0 = tbase + ch * 32 + sub * 16;
        v16bf bt = load_tile_B(kb, b * S_ + t0, E_, h * D_);
        v8f c = wmma_bf16(aq, bt, v8f_zero());
#pragma unroll
        for (int g = 0; g < 8; ++g) {
          const int rl = g + 8 * hf;
          const int row = s0 + rl;
          const int col = t0 + colL;
          const unsigned bits = maskw[(b * S_ + row) * (S_ / 32) + (col >> 5)];
          float p = 0.f;
          if ((bits >> (col & 31)) & 1u)
            p = __expf(c[g] * scale - Mrow[rl]) / Srow[rl];
          const size_t ai = ((size_t)(b * S_ + row)) * S_ + col;
          const float acc = (h == 0) ? 0.f : attn[ai];
          attn[ai] = acc + p * (1.f / H_);
          Pstage[wave][rl][sub * 16 + colL] = f32_to_bf16(p);
        }
      }
      __syncthreads();
      v16bf ap = load_tile_A(&Pstage[wave][0][0], 0, 32, 0);
      v16bf b0 = load_tile_B(vtb, (b * H_ + h) * D_ + 0,  S_, tbase + ch * 32);
      v16bf b1 = load_tile_B(vtb, (b * H_ + h) * D_ + 16, S_, tbase + ch * 32);
      c0 = wmma_bf16(ap, b0, c0);
      c1 = wmma_bf16(ap, b1, c1);
      __syncthreads();
    }
    // partial ctx reduce across waves
#pragma unroll
    for (int g = 0; g < 8; ++g) {
      cred[wave][g + 8 * hf][colL]      = c0[g];
      cred[wave][g + 8 * hf][16 + colL] = c1[g];
    }
    __syncthreads();
    for (int i = threadIdx.x; i < 16 * 32; i += 128) {
      const int r = i >> 5, cc = i & 31;
      const float v = cred[0][r][cc] + cred[1][r][cc] + cred[2][r][cc] + cred[3][r][cc];
      ctxb[((size_t)(b * S_ + s0 + r)) * E_ + h * D_ + cc] = f32_to_bf16(v);
    }
    __syncthreads();
  }
}

// ---------- launch ----------

extern "C" void kernel_launch(void* const* d_in, const int* in_sizes, int n_in,
                              void* d_out, int out_size, void* d_ws, size_t ws_size,
                              hipStream_t stream) {
  const float* x  = (const float*)d_in[0];
  const float* Wq = (const float*)d_in[1];
  const float* bq = (const float*)d_in[2];
  const float* Wk = (const float*)d_in[3];
  const float* bk = (const float*)d_in[4];
  const float* Wv = (const float*)d_in[5];
  const float* bv = (const float*)d_in[6];
  const float* Wo = (const float*)d_in[7];
  const float* bo = (const float*)d_in[8];

  float* out  = (float*)d_out;
  float* attn = out + (size_t)B_ * S_ * E_;   // [B,S,S] region, reused as score scratch

  const size_t NX = (size_t)B_ * S_ * E_;     // 2,097,152 elements
  const size_t NW = (size_t)E_ * E_;          // 65,536 elements

  unsigned short* xb   = (unsigned short*)d_ws;
  unsigned short* wqb  = xb + NX;
  unsigned short* wkb  = wqb + NW;
  unsigned short* wvb  = wkb + NW;
  unsigned short* wob  = wvb + NW;
  unsigned short* qb   = wob + NW;
  unsigned short* kbuf = qb + NX;
  unsigned short* vtb  = kbuf + NX;           // V transposed: [b*E + e][t]
  unsigned short* ctxb = vtb + NX;
  float*    thresh = (float*)(ctxb + NX);
  unsigned* maskw  = (unsigned*)(thresh + (size_t)B_ * S_);

  cvt_kernel<<<(int)((NX + 255) / 256), 256, 0, stream>>>(x,  xb,  (int)NX);
  cvt_kernel<<<(int)((NW + 255) / 256), 256, 0, stream>>>(Wq, wqb, (int)NW);
  cvt_kernel<<<(int)((NW + 255) / 256), 256, 0, stream>>>(Wk, wkb, (int)NW);
  cvt_kernel<<<(int)((NW + 255) / 256), 256, 0, stream>>>(Wv, wvb, (int)NW);
  cvt_kernel<<<(int)((NW + 255) / 256), 256, 0, stream>>>(Wo, wob, (int)NW);

  scores_kernel<<<dim3(S_ / 16, S_ / 16, B_), 32, 0, stream>>>(xb, attn);
  thresh_kernel<<<B_ * S_, 256, 0, stream>>>(attn, thresh);
  mask_kernel<<<B_ * S_, 64, 0, stream>>>(attn, thresh, maskw);

  proj_kernel<<<dim3((B_ * S_) / 16, E_ / 16), 32, 0, stream>>>(xb, wqb, bq, (void*)qb,   0);
  proj_kernel<<<dim3((B_ * S_) / 16, E_ / 16), 32, 0, stream>>>(xb, wkb, bk, (void*)kbuf, 1);
  proj_kernel<<<dim3((B_ * S_) / 16, E_ / 16), 32, 0, stream>>>(xb, wvb, bv, (void*)vtb,  2);

  attn_kernel<<<dim3(S_ / 16, B_), 128, 0, stream>>>(qb, kbuf, vtb, maskw, attn, ctxb);

  proj_kernel<<<dim3((B_ * S_) / 16, E_ / 16), 32, 0, stream>>>(ctxb, wob, bo, (void*)out, 3);
}
